// LocalAttention_21328807592567
// MI455X (gfx1250) — matrix operand order repaired and checked by reading, not compiled, not run on hardware
//
#include <hip/hip_runtime.h>
#include <math.h>

// Local (bucketed) attention for MI455X / gfx1250, wave32 + WMMA bf16.
// B=8 H=8 T=8192 E=64, BUCKETS=128 -> bucket size 64, window 128 (look_back=1).

typedef __attribute__((ext_vector_type(16))) __bf16 v16bf;
typedef __attribute__((ext_vector_type(8)))  __bf16 v8bf;
typedef __attribute__((ext_vector_type(4)))  __bf16 v4bf;
typedef __attribute__((ext_vector_type(8)))  float  v8f;
typedef __attribute__((ext_vector_type(4)))  float  v4f;

#define BQ      64    // bucket size (queries per bucket)
#define WIN     128   // key window (prev bucket + current bucket)
#define EDIM    64
#define TSEQ    8192
#define NBUCK   128
#define NBH     64    // B*H

union FragA { v16bf v; v8bf h[2]; v4bf q[4]; __bf16 b[16]; };

__global__ __launch_bounds__(128) void local_attn_wmma(
    const float* __restrict__ q, const float* __restrict__ k,
    const float* __restrict__ v, float* __restrict__ out)
{
    // bf16 staging: K rows [j][e], V transposed [e][j], probabilities [i][j]
    __shared__ __attribute__((aligned(32))) __bf16 sK [WIN * EDIM];  // 16 KB
    __shared__ __attribute__((aligned(32))) __bf16 sVt[EDIM * WIN];  // 16 KB
    __shared__ __attribute__((aligned(32))) __bf16 sP [BQ  * WIN];   // 16 KB

    const int bucket = blockIdx.x;
    const int bh     = blockIdx.y;
    const int tid    = threadIdx.x;
    const int wave   = tid >> 5;
    const int lane   = tid & 31;
    const int l16    = lane & 15;
    const int lhalf  = lane >> 4;          // 0: lanes 0-15, 1: lanes 16-31

    const size_t seq_base = (size_t)bh * TSEQ;

    // ---------------- stage K / V window into LDS (fp32 -> bf16, HW cvt) -------
    {
        const int j   = tid;                       // 128 threads = 128 window rows
        const int src = (bucket - 1) * BQ + j;     // global sequence row of key j
        const bool pad = (src < 0);                // bucket 0 backward half
        const float* kp = k + (seq_base + (size_t)(pad ? 0 : src)) * EDIM;
        const float* vp = v + (seq_base + (size_t)(pad ? 0 : src)) * EDIM;
        #pragma unroll
        for (int e = 0; e < EDIM; e += 4) {
            v4f kf = *(const v4f*)(kp + e);
            v4f vf = *(const v4f*)(vp + e);
            if (pad) { kf = (v4f){0.f,0.f,0.f,0.f}; vf = kf; }
            // K row-major: packed 4-wide bf16 store (ds_store_b64)
            *(v4bf*)&sK[j*EDIM + e] = __builtin_convertvector(kf, v4bf);
            // V transposed: per-element b16 scatter (unavoidable for transpose)
            v4bf vb = __builtin_convertvector(vf, v4bf);
            sVt[(e + 0)*WIN + j] = vb.x;
            sVt[(e + 1)*WIN + j] = vb.y;
            sVt[(e + 2)*WIN + j] = vb.z;
            sVt[(e + 3)*WIN + j] = vb.w;
        }
    }

    // ---------------- Q A-fragments straight from global -----------------------
    // 16-bit A layout (16x32): lanes 0-15 row M=l16 hold K {0..7,16..23},
    // lanes 16-31 same rows hold K {8..15,24..31}; VGPR v packs K=2v,2v+1.
    FragA aq[2];
    {
        const int qrow = bucket * BQ + wave * 16 + l16;
        const float* qp = q + (seq_base + qrow) * EDIM;
        #pragma unroll
        for (int kc = 0; kc < 2; ++kc) {
            const int e0 = kc * 32 + lhalf * 8;
            aq[kc].q[0] = __builtin_convertvector(*(const v4f*)(qp + e0),      v4bf);
            aq[kc].q[1] = __builtin_convertvector(*(const v4f*)(qp + e0 + 4),  v4bf);
            aq[kc].q[2] = __builtin_convertvector(*(const v4f*)(qp + e0 + 16), v4bf);
            aq[kc].q[3] = __builtin_convertvector(*(const v4f*)(qp + e0 + 20), v4bf);
        }
    }
    __syncthreads();

    // ---------------- scores S = Q * K^T  (8 tiles of 16x16 per wave) ----------
    // B layout (32x16): lane n=l16; lanes 0-15 hold K {0..7,16..23}, lanes 16-31 K {8..15,24..31}
    // -> each lane reads two 8-wide bf16 chunks from one K row.
    v8f sc[8];
    #pragma unroll
    for (int jt = 0; jt < 8; ++jt) {
        v8f acc = {0.f,0.f,0.f,0.f,0.f,0.f,0.f,0.f};
        #pragma unroll
        for (int kc = 0; kc < 2; ++kc) {
            FragA bk;
            bk.h[0] = *(const v8bf*)&sK[(jt*16 + l16)*EDIM + kc*32 +      lhalf*8];
            bk.h[1] = *(const v8bf*)&sK[(jt*16 + l16)*EDIM + kc*32 + 16 + lhalf*8];
            const v16bf bfrag = bk.v;
            acc = __builtin_amdgcn_wmma_f32_16x16x32_bf16(
                false, aq[kc].v, false, bfrag, (short)0, acc, false, false);
        }
        sc[jt] = acc;
    }

    // ---------------- mask + softmax in registers (C layout) -------------------
    // C layout: lane half selects M+8, VGPR r -> M = r (+8), N = l16.
    const float NEG = -3.402823466e38f;
    const bool pad0 = (bucket == 0);
    float rowmax[8], rowsum[8], rinv[8];
    #pragma unroll
    for (int r = 0; r < 8; ++r) rowmax[r] = NEG;

    #pragma unroll
    for (int jt = 0; jt < 8; ++jt) {
        const int j = jt * 16 + l16;               // window column
        #pragma unroll
        for (int r = 0; r < 8; ++r) {
            const int i = wave * 16 + lhalf * 8 + r;   // query row in bucket
            const bool masked = (j >= 64) ? ((j - 64) > i) : pad0;
            const float s = masked ? NEG : sc[jt][r];
            sc[jt][r] = s;
            rowmax[r] = fmaxf(rowmax[r], s);
        }
    }
    #pragma unroll
    for (int m = 1; m < 16; m <<= 1) {            // reduce over 16-lane half
        #pragma unroll
        for (int r = 0; r < 8; ++r)
            rowmax[r] = fmaxf(rowmax[r], __shfl_xor(rowmax[r], m, 32));
    }
    #pragma unroll
    for (int r = 0; r < 8; ++r) rowsum[r] = 0.f;
    #pragma unroll
    for (int jt = 0; jt < 8; ++jt) {
        #pragma unroll
        for (int r = 0; r < 8; ++r) {
            const float p = __expf(sc[jt][r] - rowmax[r]);  // masked -> 0
            sc[jt][r] = p;
            rowsum[r] += p;
        }
    }
    #pragma unroll
    for (int m = 1; m < 16; m <<= 1) {
        #pragma unroll
        for (int r = 0; r < 8; ++r)
            rowsum[r] += __shfl_xor(rowsum[r], m, 32);
    }
    #pragma unroll
    for (int r = 0; r < 8; ++r) rinv[r] = 1.0f / rowsum[r];

    // P -> LDS bf16 (C-layout -> row-major), consumed as A by the same wave
    #pragma unroll
    for (int jt = 0; jt < 8; ++jt) {
        #pragma unroll
        for (int r = 0; r < 8; ++r)
            sP[(wave*16 + lhalf*8 + r)*WIN + jt*16 + l16] = (__bf16)sc[jt][r];
    }
    __syncthreads();

    // ---------------- O = P * V  (4 tiles of 16x16 per wave) -------------------
    FragA ap[4];
    {
        const int prow = (wave * 16 + l16) * WIN;
        #pragma unroll
        for (int kc = 0; kc < 4; ++kc) {
            ap[kc].h[0] = *(const v8bf*)&sP[prow + kc*32 +      lhalf*8];
            ap[kc].h[1] = *(const v8bf*)&sP[prow + kc*32 + 16 + lhalf*8];
        }
    }
    #pragma unroll
    for (int nt = 0; nt < 4; ++nt) {
        v8f acc = {0.f,0.f,0.f,0.f,0.f,0.f,0.f,0.f};
        #pragma unroll
        for (int kc = 0; kc < 4; ++kc) {
            FragA bv;
            bv.h[0] = *(const v8bf*)&sVt[(nt*16 + l16)*WIN + kc*32 +      lhalf*8];
            bv.h[1] = *(const v8bf*)&sVt[(nt*16 + l16)*WIN + kc*32 + 16 + lhalf*8];
            const v16bf bfrag = bv.v;
            acc = __builtin_amdgcn_wmma_f32_16x16x32_bf16(
                false, ap[kc].v, false, bfrag, (short)0, acc, false, false);
        }
        #pragma unroll
        for (int r = 0; r < 8; ++r) {
            const int i = wave * 16 + lhalf * 8 + r;
            out[(seq_base + bucket * BQ + i) * EDIM + nt * 16 + l16] =
                acc[r] * rinv[r];
        }
    }
}

extern "C" void kernel_launch(void* const* d_in, const int* in_sizes, int n_in,
                              void* d_out, int out_size, void* d_ws, size_t ws_size,
                              hipStream_t stream) {
    const float* q = (const float*)d_in[0];
    const float* k = (const float*)d_in[1];
    const float* v = (const float*)d_in[2];
    float* o = (float*)d_out;
    dim3 grid(NBUCK, NBH);     // one workgroup per (bucket, batch*head)
    local_attn_wmma<<<grid, 128, 0, stream>>>(q, k, v, o);
}
